// RNNEncoder_12695923327217
// MI455X (gfx1250) — matrix operand order repaired and checked by reading, not compiled, not run on hardware
//
#include <hip/hip_runtime.h>

#define SEQ    4096
#define EMBED  1024
#define HIDDEN 1024

#define ROWS 16                 // hidden indices owned per persistent block
#define NWG  (HIDDEN / ROWS)    // 64 persistent blocks
#define TPB  256
#define WHT_STRIDE 17           // coprime with 64 LDS banks -> conflict-free scatter

typedef __bf16 bf16;
typedef __attribute__((ext_vector_type(16))) __bf16 v16bf;
typedef __attribute__((ext_vector_type(8)))  __bf16 v8bf;
typedef __attribute__((ext_vector_type(8)))  float  v8f;

// ---- CDNA5 async global->LDS (ASYNCcnt path), inline asm (toolchain-portable)
__device__ __forceinline__ void async_load_b128(void* lds, const void* g) {
  asm volatile("global_load_async_to_lds_b128 %0, %1, off"
               :: "v"((unsigned)(unsigned long long)lds),
                  "v"((unsigned long long)g)
               : "memory");
}
__device__ __forceinline__ void async_load_b32(void* lds, const void* g) {
  asm volatile("global_load_async_to_lds_b32 %0, %1, off"
               :: "v"((unsigned)(unsigned long long)lds),
                  "v"((unsigned long long)g)
               : "memory");
}
__device__ __forceinline__ void wait_async0() {
  asm volatile("s_wait_asynccnt 0x0" ::: "memory");
}

// ---------------- Kernel 1: embedding gather, f32 -> bf16 ----------------
__global__ void k_gather(const int* __restrict__ seq, const float* __restrict__ emb,
                         bf16* __restrict__ Xb) {
  const int t   = blockIdx.x;
  const int tok = seq[t];
  const float* src = emb + (size_t)tok * EMBED;
  bf16*        dst = Xb  + (size_t)t   * EMBED;
  for (int i = threadIdx.x; i < EMBED; i += blockDim.x)
    dst[i] = (bf16)src[i];
}

// ---------------- Kernel 2: f32 -> bf16 weight convert ----------------
__global__ void k_cvt(const float* __restrict__ src, bf16* __restrict__ dst, int n) {
  int i = blockIdx.x * blockDim.x + threadIdx.x;
  if (i < n) dst[i] = (bf16)src[i];
}

// ---------------- Kernel 3: C[m][n] = sum_k X[m][k] * W[n][k]  (WMMA bf16) --
// Wave tile 32x64: 2 A-frags x 4 B-frags -> 8 WMMAs per 32-wide k-step.
// Block = 8 waves (4 row-groups x 2 col-groups) -> 128x128 block tile.
__global__ __launch_bounds__(TPB) void k_gemm(const bf16* __restrict__ X,
                                              const bf16* __restrict__ W,
                                              float* __restrict__ C) {
  const int K      = EMBED;
  const int lane   = threadIdx.x & 31;
  const int wave   = threadIdx.x >> 5;                 // 0..7
  const int m_base = blockIdx.x * 128 + (wave >> 1) * 32;
  const int n_base = blockIdx.y * 128 + (wave & 1)  * 64;
  const int lhalf  = lane >> 4;                        // 0/1
  const int l16    = lane & 15;

  const bf16* arow0 = X + (size_t)(m_base + l16) * K + lhalf * 16;
  const bf16* arow1 = arow0 + (size_t)16 * K;
  const bf16* brow  = W + (size_t)(n_base + l16) * K + lhalf * 16;

  v8f acc[2][4];
#pragma unroll
  for (int i = 0; i < 2; ++i)
#pragma unroll
    for (int j = 0; j < 4; ++j) acc[i][j] = 0.f;

  for (int k0 = 0; k0 < K; k0 += 32) {
    __builtin_prefetch(arow0 + k0 + 128, 0, 1);   // global_prefetch_b8
    __builtin_prefetch(arow1 + k0 + 128, 0, 1);
    __builtin_prefetch(brow  + k0 + 128, 0, 1);

    v8bf a0lo = *(const v8bf*)(arow0 + k0);
    v8bf a0hi = *(const v8bf*)(arow0 + k0 + 8);
    v8bf a1lo = *(const v8bf*)(arow1 + k0);
    v8bf a1hi = *(const v8bf*)(arow1 + k0 + 8);
    v16bf a0 = __builtin_shufflevector(a0lo, a0hi, 0,1,2,3,4,5,6,7,8,9,10,11,12,13,14,15);
    v16bf a1 = __builtin_shufflevector(a1lo, a1hi, 0,1,2,3,4,5,6,7,8,9,10,11,12,13,14,15);

    v16bf b0 = *(const v16bf*)(brow + (size_t)0  * K + k0);
    v16bf b1 = *(const v16bf*)(brow + (size_t)16 * K + k0);
    v16bf b2 = *(const v16bf*)(brow + (size_t)32 * K + k0);
    v16bf b3 = *(const v16bf*)(brow + (size_t)48 * K + k0);

    acc[0][0] = __builtin_amdgcn_wmma_f32_16x16x32_bf16(false, a0, false, b0, (short)0, acc[0][0], false, false);
    acc[0][1] = __builtin_amdgcn_wmma_f32_16x16x32_bf16(false, a0, false, b1, (short)0, acc[0][1], false, false);
    acc[0][2] = __builtin_amdgcn_wmma_f32_16x16x32_bf16(false, a0, false, b2, (short)0, acc[0][2], false, false);
    acc[0][3] = __builtin_amdgcn_wmma_f32_16x16x32_bf16(false, a0, false, b3, (short)0, acc[0][3], false, false);
    acc[1][0] = __builtin_amdgcn_wmma_f32_16x16x32_bf16(false, a1, false, b0, (short)0, acc[1][0], false, false);
    acc[1][1] = __builtin_amdgcn_wmma_f32_16x16x32_bf16(false, a1, false, b1, (short)0, acc[1][1], false, false);
    acc[1][2] = __builtin_amdgcn_wmma_f32_16x16x32_bf16(false, a1, false, b2, (short)0, acc[1][2], false, false);
    acc[1][3] = __builtin_amdgcn_wmma_f32_16x16x32_bf16(false, a1, false, b3, (short)0, acc[1][3], false, false);
  }

  // C/D layout: lanes 0-15 -> M=v, N=lane; lanes 16-31 -> M=8+v, N=lane-16
#pragma unroll
  for (int i = 0; i < 2; ++i) {
    float* cbase = C + (size_t)(m_base + i * 16 + 8 * lhalf) * HIDDEN + n_base + l16;
#pragma unroll
    for (int v = 0; v < 8; ++v) {
      cbase[(size_t)v * HIDDEN + 0]  = acc[i][0][v];
      cbase[(size_t)v * HIDDEN + 16] = acc[i][1][v];
      cbase[(size_t)v * HIDDEN + 32] = acc[i][2][v];
      cbase[(size_t)v * HIDDEN + 48] = acc[i][3][v];
    }
  }
}

// ---------------- Kernel 4: init scan state ----------------
__global__ void k_init(float* __restrict__ acc3, unsigned* __restrict__ cnt) {
  if (threadIdx.x == 0) *cnt = 0u;
  for (int i = threadIdx.x; i < 3 * HIDDEN; i += blockDim.x) acc3[i] = 0.0f;
}

// ---------------- Kernel 5: persistent GRU scan, ONE barrier per step ------
__device__ __forceinline__ float sigmoidf_(float x) { return 1.0f / (1.0f + __expf(-x)); }

__device__ __forceinline__ void grid_barrier(unsigned* cnt, unsigned target) {
  __threadfence();          // flush this thread's global stores/atomics
  __syncthreads();
  if (threadIdx.x == 0) {
    __hip_atomic_fetch_add(cnt, 1u, __ATOMIC_ACQ_REL, __HIP_MEMORY_SCOPE_AGENT);
    while (__hip_atomic_load(cnt, __ATOMIC_ACQUIRE, __HIP_MEMORY_SCOPE_AGENT) < target)
      __builtin_amdgcn_s_sleep(1);
  }
  __syncthreads();
}

__global__ __launch_bounds__(TPB) void k_scan(const float* __restrict__ Wrh,
                                              const float* __restrict__ Wzh,
                                              const float* __restrict__ Wh,
                                              const float* __restrict__ xr,
                                              const float* __restrict__ xz,
                                              const float* __restrict__ xh,
                                              float* __restrict__ acc3,  // [3][HIDDEN]
                                              float* __restrict__ u2,    // [2][HIDDEN]
                                              float* __restrict__ w2,    // [2][HIDDEN]
                                              unsigned* __restrict__ cnt,
                                              float* __restrict__ out) {
  extern __shared__ float smem[];
  float* wz    = smem;                        // [ROWS][HIDDEN]  rows of Wzh
  float* wr    = wz  + ROWS * HIDDEN;         // [ROWS][HIDDEN]  rows of Wrh
  float* whT   = wr  + ROWS * HIDDEN;         // [HIDDEN][17]: whT[m*17+c] = Wh[m][b*ROWS+c]
  float* h_lds = whT + WHT_STRIDE * HIDDEN;   // [HIDDEN] full h (redundant per block)
  float* vloc  = h_lds + HIDDEN;              // [ROWS] local (r*h) slice

  const int b   = blockIdx.x;
  const int tid = threadIdx.x;

  // One-time weight parking via the async global->LDS engine (no VGPR round-trip).
  const size_t base = (size_t)b * ROWS * HIDDEN;
  for (int i = tid * 4; i < ROWS * HIDDEN; i += TPB * 4) {
    async_load_b128(wz + i, Wzh + base + i);            // contiguous 16B chunks
    async_load_b128(wr + i, Wrh + base + i);
  }
  for (int i = tid; i < ROWS * HIDDEN; i += TPB) {      // transposed column slice
    const int m = i >> 4, c = i & (ROWS - 1);
    async_load_b32(whT + m * WHT_STRIDE + c,
                   Wh + (size_t)m * HIDDEN + b * ROWS + c);
  }
  for (int i = tid; i < HIDDEN; i += TPB) h_lds[i] = 0.0f;   // h0 = 0
  wait_async0();                                        // s_wait_asynccnt 0
  __syncthreads();

  const int row = tid >> 4;            // 0..15: owned hidden index within slice
  const int seg = tid & 15;            // 0..15: 64-wide dot segment
  const int n   = b * ROWS + row;      // global hidden index
  const float* wzr  = wz + row * HIDDEN + seg * 64;
  const float* wrr  = wr + row * HIDDEN + seg * 64;
  const float* hseg = h_lds + seg * 64;

  unsigned epoch = 0;

  for (int t = 0; t < SEQ; ++t) {
    __syncthreads();                   // h_lds ready

    // ---- Gates: z_n, r_n for owned slice (full-h dots from local LDS) ----
    float az = 0.f, ar = 0.f;
#pragma unroll 8
    for (int j = 0; j < 64; ++j) {
      const float hv = hseg[j];
      az = fmaf(wzr[j], hv, az);
      ar = fmaf(wrr[j], hv, ar);
    }
#pragma unroll
    for (int m = 8; m >= 1; m >>= 1) {
      az += __shfl_xor(az, m, 16);
      ar += __shfl_xor(ar, m, 16);
    }
    float* uCur = u2 + (t & 1) * HIDDEN;
    float* wCur = w2 + (t & 1) * HIDDEN;
    if (seg == 0) {
      const float h_old = h_lds[n];
      const float zz = sigmoidf_(xz[(size_t)t * HIDDEN + n] + az);
      const float rg = sigmoidf_(xr[(size_t)t * HIDDEN + n] + ar);
      uCur[n] = zz * h_old;            // z*h
      wCur[n] = 1.0f - zz;             // (1-z)
      vloc[row] = rg * h_old;          // (r*h) on owned slice
    }
    __syncthreads();                   // vloc ready

    // ---- Column scatter: acc[:, all m] += Wh[:, slice] @ vloc ----
    float* accCur = acc3 + (t % 3) * HIDDEN;
#pragma unroll
    for (int k = 0; k < 4; ++k) {
      const int m = tid + k * TPB;
      const float* wrow = whT + m * WHT_STRIDE;   // 17-stride: bank-conflict-free
      float p = 0.f;
#pragma unroll
      for (int c = 0; c < ROWS; ++c)
        p = fmaf(wrow[c], vloc[c], p);
      atomicAdd(&accCur[m], p);        // global_atomic_add_f32, 64 adders/addr
    }

    epoch += NWG; grid_barrier(cnt, epoch);   // the ONLY grid sync per step

    // ---- Redundant full-h update (4 tanh/thread) + rotate-zero ----
    float* accZ = acc3 + ((t + 2) % 3) * HIDDEN;
    const bool last = (t == SEQ - 1);
#pragma unroll
    for (int k = 0; k < 4; ++k) {
      const int m = tid + k * TPB;
      const float hn = uCur[m] + wCur[m] * tanhf(xh[(size_t)t * HIDDEN + m] + accCur[m]);
      h_lds[m] = hn;
      accZ[m] = 0.0f;                  // race-free: consumed 1 step ago, reused 2 ahead
      if (last && b == 0) out[m] = hn;
    }
  }
}

// ---------------- host launcher ----------------
extern "C" void kernel_launch(void* const* d_in, const int* in_sizes, int n_in,
                              void* d_out, int out_size, void* d_ws, size_t ws_size,
                              hipStream_t stream) {
  (void)in_sizes; (void)n_in; (void)out_size; (void)ws_size;
  const int*   seq = (const int*)d_in[0];
  const float* emb = (const float*)d_in[1];
  const float* Wrx = (const float*)d_in[2];
  const float* Wrh = (const float*)d_in[3];
  const float* Wzx = (const float*)d_in[4];
  const float* Wzh = (const float*)d_in[5];
  const float* Wx  = (const float*)d_in[6];
  const float* Wh  = (const float*)d_in[7];
  float* out = (float*)d_out;

  char* p = (char*)d_ws;
  auto carve = [&](size_t bytes) -> char* {
    char* r = p;
    p += (bytes + 255) & ~(size_t)255;
    return r;
  };
  bf16* Xb    = (bf16*)carve((size_t)SEQ * EMBED * sizeof(bf16));
  bf16* Wrxb  = (bf16*)carve((size_t)HIDDEN * EMBED * sizeof(bf16));
  bf16* Wzxb  = (bf16*)carve((size_t)HIDDEN * EMBED * sizeof(bf16));
  bf16* Wxb   = (bf16*)carve((size_t)HIDDEN * EMBED * sizeof(bf16));
  float* xr   = (float*)carve((size_t)SEQ * HIDDEN * sizeof(float));
  float* xz   = (float*)carve((size_t)SEQ * HIDDEN * sizeof(float));
  float* xh   = (float*)carve((size_t)SEQ * HIDDEN * sizeof(float));
  float* acc3 = (float*)carve(3 * HIDDEN * sizeof(float));
  float* u2   = (float*)carve(2 * HIDDEN * sizeof(float));
  float* w2   = (float*)carve(2 * HIDDEN * sizeof(float));
  unsigned* cnt = (unsigned*)carve(256);

  // Phase 1: gather + bf16 conversion
  k_gather<<<SEQ, 256, 0, stream>>>(seq, emb, Xb);
  const int nW = HIDDEN * EMBED;
  k_cvt<<<(nW + 255) / 256, 256, 0, stream>>>(Wrx, Wrxb, nW);
  k_cvt<<<(nW + 255) / 256, 256, 0, stream>>>(Wzx, Wzxb, nW);
  k_cvt<<<(nW + 255) / 256, 256, 0, stream>>>(Wx,  Wxb,  nW);

  // Phase 2: input-side GEMMs on the WMMA path (32x64 wave tiles)
  dim3 gg(SEQ / 128, HIDDEN / 128);
  k_gemm<<<gg, TPB, 0, stream>>>(Xb, Wrxb, xr);
  k_gemm<<<gg, TPB, 0, stream>>>(Xb, Wzxb, xz);
  k_gemm<<<gg, TPB, 0, stream>>>(Xb, Wxb,  xh);

  // Phase 3: persistent LDS-resident GRU scan, one grid barrier per step
  k_init<<<1, 256, 0, stream>>>(acc3, cnt);
  const size_t smem_bytes =
      (size_t)(2 * ROWS * HIDDEN + WHT_STRIDE * HIDDEN + HIDDEN + ROWS) * sizeof(float);
  (void)hipFuncSetAttribute((const void*)k_scan,
                            hipFuncAttributeMaxDynamicSharedMemorySize,
                            (int)smem_bytes);
  k_scan<<<NWG, TPB, smem_bytes, stream>>>(Wrh, Wzh, Wh, xr, xz, xh,
                                           acc3, u2, w2, cnt, out);
}
